// EquiReact_76295799046683
// MI455X (gfx1250) — matrix operand order repaired and checked by reading, not compiled, not run on hardware
//
#include <hip/hip_runtime.h>
#include <math.h>

// ---------------------------------------------------------------------------
// EquiReact-style equivariant GNN, fused for MI455X (gfx1250, wave32, WMMA).
//
// Core design: the per-edge weight-generator MLPs (fc0: 48->48->512,
// fc1: 48->48->1536 over 100k edges, ~21 GFLOP) are done with
// v_wmma_f32_16x16x32_f16, one wave per 16-edge tile, and the generated
// weights are consumed *in registers* by the tensor-product contraction +
// atomic scatter. This avoids ever materializing the 614 MB `w` tensor in
// HBM (which would dominate runtime at 23.3 TB/s).
// ---------------------------------------------------------------------------

typedef __attribute__((ext_vector_type(16))) _Float16 v16h;
typedef __attribute__((ext_vector_type(8)))  float    v8f;

#define N_NODES   10000
#define N_EDGES   100000
#define N_TILES   (N_EDGES / 16)   // 6250, exact
#define N_GRAPHS  64
#define DIST_DIM  32
#define MAX_Rf    10.0f

#define SQ3f  1.7320508075688772f
#define S75   2.7386127875258306f   // sqrt(7.5)
#define R2c   0.7071067811865476f   // 1/sqrt(2)
#define R6c   0.4082482904638630f   // 1/sqrt(6)
#define C32   0.17677669529663687f  // 1/sqrt(32)
#define C32_3 0.10206207261596575f  // 1/sqrt(32)/sqrt(3)
#define K13   0.14433756729740643f  // 0.25/sqrt(3)
#define K5c   0.11180339887498948f  // 0.25/sqrt(5)

__device__ __forceinline__ float frelu(float x) { return x > 0.f ? x : 0.f; }

#define WMMA(a, b, c) \
  __builtin_amdgcn_wmma_f32_16x16x32_f16(false, (a), false, (b), (short)0, (c), false, false)

// A-matrix (16x32 f16) fragment from an LDS tile stored [M][64] row-major.
// Layout per ISA 7.12.2: lane M = lane&15; VGPR v<4: K = 2v(+8 hi half),
// v>=4: K = 16+2(v-4)(+8 hi half); two f16 per VGPR are K, K+1.
__device__ __forceinline__ v16h load_a_frag(const _Float16* s, int M, int kbase, int hioff) {
  v16h a;
#pragma unroll
  for (int v = 0; v < 8; ++v) {
    int k = kbase + hioff + ((v < 4) ? (2 * v) : (16 + 2 * (v - 4)));
    a[2 * v]     = s[M * 64 + k];
    a[2 * v + 1] = s[M * 64 + k + 1];
  }
  return a;
}

// ---------------------------------------------------------------------------
__global__ void k_zero(float* __restrict__ p, int n) {
  int i = blockIdx.x * blockDim.x + threadIdx.x;
  if (i < n) p[i] = 0.f;
}

// node embedding: x(64) -> relu(16) -> 16
__global__ void k_node_emb(const float* __restrict__ x,
                           const float* __restrict__ w1, const float* __restrict__ b1,
                           const float* __restrict__ w2, const float* __restrict__ b2,
                           float* __restrict__ h0) {
  int n = blockIdx.x * blockDim.x + threadIdx.x;
  if (n >= N_NODES) return;
  float hid[16];
#pragma unroll
  for (int j = 0; j < 16; ++j) {
    float s = b1[j];
    for (int i = 0; i < 64; ++i) s += x[n * 64 + i] * w1[i * 16 + j];
    hid[j] = frelu(s);
  }
#pragma unroll
  for (int o = 0; o < 16; ++o) {
    float s = b2[o];
    for (int j = 0; j < 16; ++j) s += hid[j] * w2[j * 16 + o];
    h0[n * 16 + o] = s;
  }
}

// per-edge: sph harmonics, gaussian RBF, edge-embedding MLP, degree count
__global__ void k_edge_pre(const int* __restrict__ ei, const float* __restrict__ pos,
                           const float* __restrict__ w1, const float* __restrict__ b1,
                           const float* __restrict__ w2, const float* __restrict__ b2,
                           float* __restrict__ eemb, float* __restrict__ sh1b,
                           float* __restrict__ sh2b, float* __restrict__ cnt) {
  int e = blockIdx.x * blockDim.x + threadIdx.x;
  if (e >= N_EDGES) return;
  int s = ei[e], d = ei[N_EDGES + e];
  float ex = pos[d * 3 + 0] - pos[s * 3 + 0];
  float ey = pos[d * 3 + 1] - pos[s * 3 + 1];
  float ez = pos[d * 3 + 2] - pos[s * 3 + 2];
  float dd = sqrtf(ex * ex + ey * ey + ez * ez);
  float inv = 1.f / dd;
  float u0 = ex * inv, u1 = ey * inv, u2 = ez * inv;
  sh1b[e * 3 + 0] = SQ3f * u0;
  sh1b[e * 3 + 1] = SQ3f * u1;
  sh1b[e * 3 + 2] = SQ3f * u2;
  sh2b[e * 5 + 0] = S75 * 2.f * R2c * u0 * u1;
  sh2b[e * 5 + 1] = S75 * 2.f * R2c * u0 * u2;
  sh2b[e * 5 + 2] = S75 * 2.f * R2c * u1 * u2;
  sh2b[e * 5 + 3] = S75 * R2c * (u0 * u0 - u1 * u1);
  sh2b[e * 5 + 4] = S75 * R6c * (2.f * u2 * u2 - u0 * u0 - u1 * u1);
  // gaussian smearing
  float g[DIST_DIM];
  const float step = MAX_Rf / (float)(DIST_DIM - 1);
  const float coeff = -0.5f / (step * step);
  for (int m = 0; m < DIST_DIM; ++m) {
    float t = dd - step * (float)m;
    g[m] = expf(coeff * t * t);
  }
  float hid[16];
#pragma unroll
  for (int j = 0; j < 16; ++j) {
    float ss = b1[j];
    for (int i = 0; i < DIST_DIM; ++i) ss += g[i] * w1[i * 16 + j];
    hid[j] = frelu(ss);
  }
#pragma unroll
  for (int o = 0; o < 16; ++o) {
    float ss = b2[o];
    for (int j = 0; j < 16; ++j) ss += hid[j] * w2[j * 16 + o];
    eemb[e * 16 + o] = ss;
  }
  atomicAdd(cnt + d, 1.f);
}

// Pre-swizzle a (Kdim x Ncols) f32 weight into f16 WMMA B-fragments.
// B 32x16 layout (matches A fragment): element i<8 -> K=8h+i, i>=8 ->
// K=16+8h+(i-8), h = lane/16. K padded to 64.
__global__ void k_swz(const float* __restrict__ W, int Kdim, int Ncols, int ntiles,
                      _Float16* __restrict__ out) {
  int idx = blockIdx.x * blockDim.x + threadIdx.x;
  int total = ntiles * 2 * 32 * 16;
  if (idx >= total) return;
  int i    = idx & 15;
  int lane = (idx >> 4) & 31;
  int ks   = (idx >> 9) & 1;
  int t    = idx >> 10;
  int N = lane & 15;
  int K = ks * 32 + ((lane >= 16) ? 8 : 0) + ((i < 8) ? i : (16 + i - 8));
  float val = (K < Kdim) ? W[K * Ncols + t * 16 + N] : 0.f;
  out[idx] = (_Float16)val;
}

// ---------------------------------------------------------------------------
// Fused fc0 MLP (48->48 relu ->512) + tp_layer0 + atomic scatter. 1 wave/tile.
__global__ __launch_bounds__(32) void k_fc0_tp0(
    const int* __restrict__ ei, const float* __restrict__ h0,
    const float* __restrict__ eemb, const float* __restrict__ sh1b,
    const float* __restrict__ b1, const float* __restrict__ b2,
    const _Float16* __restrict__ bw1, const _Float16* __restrict__ bw2,
    float* __restrict__ upd) {
  __shared__ _Float16 sea[16 * 64];
  __shared__ _Float16 sm[16 * 64];
  __shared__ float sxs[16 * 16];
  __shared__ float ssh1[16 * 3];
  __shared__ int sdst[16];
  const int lane = threadIdx.x;
  const int tile = blockIdx.x;
  if (lane < 16) {
    int e = tile * 16 + lane;
    int s = ei[e], d = ei[N_EDGES + e];
    sdst[lane] = d;
    for (int i = 0; i < 16; ++i) sea[lane * 64 + i]      = (_Float16)eemb[e * 16 + i];
    for (int i = 0; i < 16; ++i) sea[lane * 64 + 16 + i] = (_Float16)h0[d * 16 + i];
    for (int i = 0; i < 16; ++i) {
      float v = h0[s * 16 + i];
      sea[lane * 64 + 32 + i] = (_Float16)v;
      sxs[lane * 16 + i] = v;
    }
    for (int i = 48; i < 64; ++i) sea[lane * 64 + i] = (_Float16)0.f;
    for (int k = 0; k < 3; ++k) ssh1[lane * 3 + k] = sh1b[e * 3 + k];
    for (int i = 0; i < 64; ++i) sm[lane * 64 + i] = (_Float16)0.f;
  }
  __syncthreads();
  const int M = lane & 15;
  const int hioff = (lane >> 4) * 8;   // K offset for A fragments
  const int mhi = (lane >> 4) * 8;     // M offset for C rows
  const int N = lane & 15;
  v16h a0 = load_a_frag(sea, M, 0, hioff);
  v16h a1 = load_a_frag(sea, M, 32, hioff);
  // GEMM1 (48 cols = 3 N-tiles), relu, restage as f16 A-operand
  for (int t = 0; t < 3; ++t) {
    v16h bb0 = *reinterpret_cast<const v16h*>(bw1 + ((t * 2 + 0) * 32 + lane) * 16);
    v16h bb1 = *reinterpret_cast<const v16h*>(bw1 + ((t * 2 + 1) * 32 + lane) * 16);
    v8f c = {};
    c = WMMA(a0, bb0, c);
    c = WMMA(a1, bb1, c);
    float bias = b1[t * 16 + N];
#pragma unroll
    for (int r = 0; r < 8; ++r)
      sm[(r + mhi) * 64 + t * 16 + N] = (_Float16)frelu(c[r] + bias);
  }
  __syncthreads();
  v16h a20 = load_a_frag(sm, M, 0, hioff);
  v16h a21 = load_a_frag(sm, M, 32, hioff);
  float acc0[8], acc1[8];
#pragma unroll
  for (int r = 0; r < 8; ++r) { acc0[r] = 0.f; acc1[r] = 0.f; }
  // GEMM2 (512 cols = 32 N-tiles) consumed in registers by tp_layer0
  for (int t = 0; t < 32; ++t) {
    v16h bb0 = *reinterpret_cast<const v16h*>(bw2 + ((t * 2 + 0) * 32 + lane) * 16);
    v16h bb1 = *reinterpret_cast<const v16h*>(bw2 + ((t * 2 + 1) * 32 + lane) * 16);
    v8f c = {};
    c = WMMA(a20, bb0, c);
    c = WMMA(a21, bb1, c);
    float bias = b2[t * 16 + N];
    int u = t & 15;
    if (t < 16) {
#pragma unroll
      for (int r = 0; r < 8; ++r)
        acc0[r] += 0.25f * sxs[(r + mhi) * 16 + u] * (c[r] + bias);
    } else {
#pragma unroll
      for (int r = 0; r < 8; ++r)
        acc1[r] += 0.25f * sxs[(r + mhi) * 16 + u] * (c[r] + bias);
    }
  }
#pragma unroll
  for (int r = 0; r < 8; ++r) {
    int Mr = r + mhi;
    float* base = upd + (long)sdst[Mr] * 64;
    atomicAdd(base + N, acc0[r]);
    for (int k = 0; k < 3; ++k)
      atomicAdd(base + 16 + N * 3 + k, acc1[r] * ssh1[Mr * 3 + k]);
  }
}

// h1 = pad(h0, 64) + upd0/count
__global__ void k_upd0(const float* __restrict__ h0, const float* __restrict__ upd0,
                       const float* __restrict__ cnt, float* __restrict__ h1) {
  int n = blockIdx.x * blockDim.x + threadIdx.x;
  if (n >= N_NODES) return;
  float c = fmaxf(cnt[n], 1.f);
  for (int j = 0; j < 64; ++j) {
    float u = upd0[n * 64 + j] / c;
    h1[n * 64 + j] = ((j < 16) ? h0[n * 16 + j] : 0.f) + u;
  }
}

// ---------------------------------------------------------------------------
// Fused fc1 MLP (48->48 relu ->1536) + tp_layer1 + atomic scatter. 1 wave/tile.
__global__ __launch_bounds__(32) void k_fc1_tp1(
    const int* __restrict__ ei, const float* __restrict__ h1,
    const float* __restrict__ eemb, const float* __restrict__ sh1b,
    const float* __restrict__ sh2b,
    const float* __restrict__ b1, const float* __restrict__ b2,
    const _Float16* __restrict__ bw1, const _Float16* __restrict__ bw2,
    float* __restrict__ upd) {
  __shared__ _Float16 sea[16 * 64];
  __shared__ _Float16 sm[16 * 64];
  __shared__ float s0f[16 * 16];
  __shared__ float v1f[16 * 16 * 3];
  __shared__ float af[16 * 16];
  __shared__ float crf[16 * 16 * 3];
  __shared__ float t5f[16 * 16 * 3];
  __shared__ float ssh1[16 * 3];
  __shared__ int sdst[16];
  const int lane = threadIdx.x;
  const int tile = blockIdx.x;
  if (lane < 16) {
    int e = tile * 16 + lane;
    int s = ei[e], d = ei[N_EDGES + e];
    sdst[lane] = d;
    for (int i = 0; i < 16; ++i) sea[lane * 64 + i]      = (_Float16)eemb[e * 16 + i];
    for (int i = 0; i < 16; ++i) sea[lane * 64 + 16 + i] = (_Float16)h1[d * 64 + i];
    for (int i = 0; i < 16; ++i) {
      float v = h1[s * 64 + i];
      sea[lane * 64 + 32 + i] = (_Float16)v;
      s0f[lane * 16 + i] = v;
    }
    for (int i = 48; i < 64; ++i) sea[lane * 64 + i] = (_Float16)0.f;
    for (int i = 0; i < 64; ++i) sm[lane * 64 + i] = (_Float16)0.f;
    float sx = sh1b[e * 3 + 0], sy = sh1b[e * 3 + 1], sz = sh1b[e * 3 + 2];
    ssh1[lane * 3 + 0] = sx; ssh1[lane * 3 + 1] = sy; ssh1[lane * 3 + 2] = sz;
    float q0 = sh2b[e * 5 + 0], q1 = sh2b[e * 5 + 1], q2 = sh2b[e * 5 + 2];
    float q3 = sh2b[e * 5 + 3], q4 = sh2b[e * 5 + 4];
    // sh2M[i][k] = sum_j sh2[j] * M2[j,i,k]
    float m00 =  q3 * R2c - q4 * R6c, m01 = q0 * R2c,              m02 = q1 * R2c;
    float m10 =  q0 * R2c,            m11 = -q3 * R2c - q4 * R6c,  m12 = q2 * R2c;
    float m20 =  q1 * R2c,            m21 = q2 * R2c,              m22 = 2.f * q4 * R6c;
    for (int u = 0; u < 16; ++u) {
      float vx = h1[s * 64 + 16 + u * 3 + 0];
      float vy = h1[s * 64 + 16 + u * 3 + 1];
      float vz = h1[s * 64 + 16 + u * 3 + 2];
      int b = (lane * 16 + u) * 3;
      v1f[b + 0] = vx; v1f[b + 1] = vy; v1f[b + 2] = vz;
      af[lane * 16 + u] = vx * sx + vy * sy + vz * sz;
      crf[b + 0] = vy * sz - vz * sy;
      crf[b + 1] = vz * sx - vx * sz;
      crf[b + 2] = vx * sy - vy * sx;
      t5f[b + 0] = vx * m00 + vy * m10 + vz * m20;
      t5f[b + 1] = vx * m01 + vy * m11 + vz * m21;
      t5f[b + 2] = vx * m02 + vy * m12 + vz * m22;
    }
  }
  __syncthreads();
  const int M = lane & 15;
  const int hioff = (lane >> 4) * 8;
  const int mhi = (lane >> 4) * 8;
  const int N = lane & 15;
  v16h a0 = load_a_frag(sea, M, 0, hioff);
  v16h a1 = load_a_frag(sea, M, 32, hioff);
  for (int t = 0; t < 3; ++t) {
    v16h bb0 = *reinterpret_cast<const v16h*>(bw1 + ((t * 2 + 0) * 32 + lane) * 16);
    v16h bb1 = *reinterpret_cast<const v16h*>(bw1 + ((t * 2 + 1) * 32 + lane) * 16);
    v8f c = {};
    c = WMMA(a0, bb0, c);
    c = WMMA(a1, bb1, c);
    float bias = b1[t * 16 + N];
#pragma unroll
    for (int r = 0; r < 8; ++r)
      sm[(r + mhi) * 64 + t * 16 + N] = (_Float16)frelu(c[r] + bias);
  }
  __syncthreads();
  v16h a20 = load_a_frag(sm, M, 0, hioff);
  v16h a21 = load_a_frag(sm, M, 32, hioff);
  float o0e[8], accT[8], o1o[3][8], o1e[3][8];
#pragma unroll
  for (int r = 0; r < 8; ++r) {
    o0e[r] = 0.f; accT[r] = 0.f;
    for (int k = 0; k < 3; ++k) { o1o[k][r] = 0.f; o1e[k][r] = 0.f; }
  }
  // GEMM2 (1536 cols = 96 N-tiles), block bi = which W[0..5], u = input channel
  for (int t = 0; t < 96; ++t) {
    v16h bb0 = *reinterpret_cast<const v16h*>(bw2 + ((t * 2 + 0) * 32 + lane) * 16);
    v16h bb1 = *reinterpret_cast<const v16h*>(bw2 + ((t * 2 + 1) * 32 + lane) * 16);
    v8f c = {};
    c = WMMA(a20, bb0, c);
    c = WMMA(a21, bb1, c);
    float bias = b2[t * 16 + N];
    int bi = t >> 4;
    int u = t & 15;
#pragma unroll
    for (int r = 0; r < 8; ++r) {
      int Mu = (r + mhi) * 16 + u;
      float w_ = c[r] + bias;
      if (bi == 0) {
        o0e[r] += C32 * s0f[Mu] * w_;
      } else if (bi == 1) {
        accT[r] += K13 * s0f[Mu] * w_;
      } else if (bi == 2) {
        for (int k = 0; k < 3; ++k) o1o[k][r] += K13 * v1f[Mu * 3 + k] * w_;
      } else if (bi == 3) {
        o0e[r] += C32_3 * af[Mu] * w_;
      } else if (bi == 4) {
        for (int k = 0; k < 3; ++k) o1e[k][r] += C32 * crf[Mu * 3 + k] * w_;
      } else {
        for (int k = 0; k < 3; ++k) o1o[k][r] += K5c * t5f[Mu * 3 + k] * w_;
      }
    }
  }
#pragma unroll
  for (int r = 0; r < 8; ++r) {
    int Mr = r + mhi;
    float* base = upd + (long)sdst[Mr] * 112;
    atomicAdd(base + N, o0e[r]);
    for (int k = 0; k < 3; ++k) {
      atomicAdd(base + 16 + N * 3 + k, o1o[k][r] + accT[r] * ssh1[Mr * 3 + k]);
      atomicAdd(base + 64 + N * 3 + k, o1e[k][r]);
    }
  }
}

// final: h2_16 = h1[:,:16] + upd1[:,:16]/cnt; score MLP; segment_sum over batch
__global__ void k_final(const float* __restrict__ h1, const float* __restrict__ upd1,
                        const float* __restrict__ cnt, const int* __restrict__ batch,
                        const float* __restrict__ w1, const float* __restrict__ b1,
                        const float* __restrict__ w2, const float* __restrict__ b2,
                        const float* __restrict__ w3, const float* __restrict__ b3,
                        float* __restrict__ out) {
  int n = blockIdx.x * blockDim.x + threadIdx.x;
  if (n >= N_NODES) return;
  float c = fmaxf(cnt[n], 1.f);
  float h16[16];
#pragma unroll
  for (int j = 0; j < 16; ++j) h16[j] = h1[n * 64 + j] + upd1[n * 112 + j] / c;
  float hA[32];
#pragma unroll
  for (int o = 0; o < 32; ++o) {
    float s = b1[o];
    for (int j = 0; j < 16; ++j) s += h16[j] * w1[j * 32 + o];
    hA[o] = frelu(s);
  }
  float hB[16];
#pragma unroll
  for (int o = 0; o < 16; ++o) {
    float s = b2[o];
    for (int j = 0; j < 32; ++j) s += hA[j] * w2[j * 16 + o];
    hB[o] = frelu(s);
  }
  float sc = b3[0];
#pragma unroll
  for (int j = 0; j < 16; ++j) sc += hB[j] * w3[j];
  atomicAdd(out + batch[n], sc);
}

// ---------------------------------------------------------------------------
extern "C" void kernel_launch(void* const* d_in, const int* in_sizes, int n_in,
                              void* d_out, int out_size, void* d_ws, size_t ws_size,
                              hipStream_t stream) {
  (void)in_sizes; (void)n_in; (void)ws_size;
  // setup_inputs dict order: x, pos, edge_index, batch, params (jax pytree:
  // dict keys sorted -> edge_emb, fc0, fc1, node_emb, score; each [(W,b),..])
  const float* x     = (const float*)d_in[0];
  const float* pos   = (const float*)d_in[1];
  const int*   ei    = (const int*)d_in[2];
  const int*   batch = (const int*)d_in[3];
  const float* ee_w1 = (const float*)d_in[4];
  const float* ee_b1 = (const float*)d_in[5];
  const float* ee_w2 = (const float*)d_in[6];
  const float* ee_b2 = (const float*)d_in[7];
  const float* f0_w1 = (const float*)d_in[8];
  const float* f0_b1 = (const float*)d_in[9];
  const float* f0_w2 = (const float*)d_in[10];
  const float* f0_b2 = (const float*)d_in[11];
  const float* f1_w1 = (const float*)d_in[12];
  const float* f1_b1 = (const float*)d_in[13];
  const float* f1_w2 = (const float*)d_in[14];
  const float* f1_b2 = (const float*)d_in[15];
  const float* ne_w1 = (const float*)d_in[16];
  const float* ne_b1 = (const float*)d_in[17];
  const float* ne_w2 = (const float*)d_in[18];
  const float* ne_b2 = (const float*)d_in[19];
  const float* sc_w1 = (const float*)d_in[20];
  const float* sc_b1 = (const float*)d_in[21];
  const float* sc_w2 = (const float*)d_in[22];
  const float* sc_b2 = (const float*)d_in[23];
  const float* sc_w3 = (const float*)d_in[24];
  const float* sc_b3 = (const float*)d_in[25];
  float* out = (float*)d_out;

  // workspace layout (float units); ~20.2 MB total
  float* ws    = (float*)d_ws;
  float* upd0  = ws;                    // 640000   (zeroed)
  float* upd1  = ws + 640000;           // 1120000  (zeroed)
  float* cnt   = ws + 1760000;          // 10000    (zeroed)
  float* h0    = ws + 1770016;          // 160000
  float* h1    = h0 + 160000;           // 640000
  float* eembB = h1 + 640000;           // 1600000
  float* sh1B  = eembB + 1600000;       // 300000
  float* sh2B  = sh1B + 300000;         // 500000
  _Float16* hb = (_Float16*)(sh2B + 500000);  // 32B-aligned
  _Float16* bs_f0_1 = hb;               // 3 tiles  * 1024 f16
  _Float16* bs_f0_2 = hb + 3072;        // 32 tiles * 1024 f16
  _Float16* bs_f1_1 = hb + 35840;       // 3 tiles
  _Float16* bs_f1_2 = hb + 38912;       // 96 tiles * 1024 f16

  k_zero<<<(1770000 + 255) / 256, 256, 0, stream>>>(ws, 1770000);
  k_zero<<<1, 64, 0, stream>>>(out, out_size);

  k_node_emb<<<(N_NODES + 127) / 128, 128, 0, stream>>>(x, ne_w1, ne_b1, ne_w2, ne_b2, h0);
  k_edge_pre<<<(N_EDGES + 127) / 128, 128, 0, stream>>>(ei, pos, ee_w1, ee_b1, ee_w2, ee_b2,
                                                        eembB, sh1B, sh2B, cnt);
  k_swz<<<(3 * 1024 + 255) / 256, 256, 0, stream>>>(f0_w1, 48, 48, 3, bs_f0_1);
  k_swz<<<(32 * 1024 + 255) / 256, 256, 0, stream>>>(f0_w2, 48, 512, 32, bs_f0_2);
  k_swz<<<(3 * 1024 + 255) / 256, 256, 0, stream>>>(f1_w1, 48, 48, 3, bs_f1_1);
  k_swz<<<(96 * 1024 + 255) / 256, 256, 0, stream>>>(f1_w2, 48, 1536, 96, bs_f1_2);

  k_fc0_tp0<<<N_TILES, 32, 0, stream>>>(ei, h0, eembB, sh1B, f0_b1, f0_b2,
                                        bs_f0_1, bs_f0_2, upd0);
  k_upd0<<<(N_NODES + 127) / 128, 128, 0, stream>>>(h0, upd0, cnt, h1);
  k_fc1_tp1<<<N_TILES, 32, 0, stream>>>(ei, h1, eembB, sh1B, sh2B, f1_b1, f1_b2,
                                        bs_f1_1, bs_f1_2, upd1);
  k_final<<<(N_NODES + 127) / 128, 128, 0, stream>>>(h1, upd1, cnt, batch,
                                                     sc_w1, sc_b1, sc_w2, sc_b2,
                                                     sc_w3, sc_b3, out);
}
